// FeatherNet_24953759990119
// MI455X (gfx1250) — matrix operand order repaired and checked by reading, not compiled, not run on hardware
//
#include <hip/hip_runtime.h>
#include <stdint.h>

// ---------------------------------------------------------------------------
// Types
// ---------------------------------------------------------------------------
typedef __attribute__((ext_vector_type(16))) __bf16       v16bf;
typedef __attribute__((ext_vector_type(8)))  float        v8f;
typedef __attribute__((ext_vector_type(4)))  unsigned int v4u;
typedef __attribute__((ext_vector_type(8)))  int          v8i;
typedef __attribute__((ext_vector_type(4)))  int          v4i;

#define TDM_OK (__has_builtin(__builtin_amdgcn_tensor_load_to_lds) && \
                __has_builtin(__builtin_amdgcn_s_wait_tensorcnt))

// LDS tile geometry: 128 rows x 32 bf16, padded to 40-elem (80 B) row stride
// -> lane l's b128 fragment reads start at bank 20*l mod 64: conflict-free.
#define ROW_ELEMS   40            // 32 data + 4-DWORD pad
#define TILE_ELEMS  (128 * ROW_ELEMS)

__device__ __forceinline__ uint16_t f2bf(float f) {
  uint32_t u = __float_as_uint(f);
  uint32_t r = (u + 0x7FFFu + ((u >> 16) & 1u)) >> 16;   // round-nearest-even
  return (uint16_t)r;
}
__device__ __forceinline__ float bf2f(uint16_t h) {
  return __uint_as_float(((uint32_t)h) << 16);
}
__device__ __forceinline__ unsigned lds_byte_off(const void* p) {
  // generic shared-aperture pointer: low 32 bits are the LDS byte offset
  return (unsigned)(uintptr_t)p;
}

union Frag {
  v16bf v;
  uint4 q[2];
};

// ---------------------------------------------------------------------------
// TDM 2D tile load: [128 rows x 32 cols] of bf16 from a row-major tensor
// (row stride = lda elements) into LDS with 16 B padding after each 64 B row
// (pad_interval = 16 DWORDs, pad_amount = 4 DWORDs) -> 80 B LDS row stride.
// 6-arg builtin: (v4u g0, v8i g1, v4i g2, v4i g3, v8i g4, cpol)
// ---------------------------------------------------------------------------
#if TDM_OK
__device__ __forceinline__ void tdm_load_tile_128x32(unsigned ldsOff,
                                                     const uint16_t* gptr,
                                                     unsigned lda_elems,
                                                     unsigned rows) {
  unsigned long long ga = (unsigned long long)(uintptr_t)gptr;
  v4u g0;
  g0[0] = 1u;                                            // count=1, no gather
  g0[1] = ldsOff;                                        // lds_addr (bytes)
  g0[2] = (unsigned)(ga & 0xFFFFFFFFull);                // global_addr[31:0]
  g0[3] = (unsigned)((ga >> 32) & 0x1FFFFFFull) | (2u << 30); // addr[56:32], type=2
  v8i g1;
  g1[0] = (int)((1u << 16) |                             // data_size=1 (2 bytes)
                (1u << 20) |                             // pad_enable
                (3u << 22) |                             // pad_interval: 16 DWORDs
                (3u << 25));                             // pad_amount:   4 DWORDs
  g1[1] = (int)((lda_elems & 0xFFFFu) << 16);            // tensor_dim0[15:0]
  g1[2] = (int)((lda_elems >> 16) | ((rows & 0xFFFFu) << 16)); // d0 hi | d1 lo
  g1[3] = (int)((rows >> 16) | (32u << 16));             // d1 hi | tile_dim0=32
  g1[4] = (int)(128u);                                   // tile_dim1=128, tile_dim2=0
  g1[5] = (int)lda_elems;                                // tensor_dim0_stride[31:0]
  g1[6] = 0;                                             // stride0 hi | stride1 lo
  g1[7] = 0;
  v4i z4 = {0, 0, 0, 0};
  v8i z8 = {0, 0, 0, 0, 0, 0, 0, 0};
  __builtin_amdgcn_tensor_load_to_lds(g0, g1, z4, z4, z8, 0);
}
#endif

// ---------------------------------------------------------------------------
// C[M,N] = act( A[M,K] @ B[N,K]^T + bias )     (bf16 inputs, f32 accumulate)
//   A row-major (lda elems/row, Mp padded rows: multiple of 128)
//   B row-major (ldb elems/row, Np padded rows: multiple of 128)
//   K multiple of 32.  Block tile 128x128, 8 waves of 64x32 each.
// ---------------------------------------------------------------------------
__global__ __launch_bounds__(256)
void gemm_bf16_wmma(const uint16_t* __restrict__ A, int lda, int Mp,
                    const uint16_t* __restrict__ B, int ldb, int Np,
                    void* __restrict__ C, int ldc, int M, int N, int K,
                    const uint16_t* __restrict__ bias, int doRelu, int outBf16) {
  __shared__ uint16_t As[2][TILE_ELEMS];
  __shared__ uint16_t Bs[2][TILE_ELEMS];

  const int tid  = threadIdx.x;
  const int lane = tid & 31;
  const int wave = tid >> 5;
  const int wm   = wave >> 2;            // 0..1  -> 64-row strip
  const int wn   = wave & 3;             // 0..3  -> 32-col strip
  const int row0 = blockIdx.y * 128;
  const int col0 = blockIdx.x * 128;
  const int NK   = K >> 5;
  const int q    = lane >> 4;            // half-wave selector
  const int l16  = lane & 15;

  v8f acc[4][2];
#pragma unroll
  for (int i = 0; i < 4; ++i)
#pragma unroll
    for (int j = 0; j < 2; ++j)
#pragma unroll
      for (int e = 0; e < 8; ++e) acc[i][j][e] = 0.0f;

#if TDM_OK
  const bool ldr = (wave == 0);
  if (ldr) {
    tdm_load_tile_128x32(lds_byte_off(&As[0][0]), A + (size_t)row0 * lda,
                         (unsigned)lda, (unsigned)Mp);
    tdm_load_tile_128x32(lds_byte_off(&Bs[0][0]), B + (size_t)col0 * ldb,
                         (unsigned)ldb, (unsigned)Np);
  }
#endif

  for (int kt = 0; kt < NK; ++kt) {
    const int cur = kt & 1;

#if TDM_OK
    if (ldr) {
      if (kt + 1 < NK) {
        const int k1 = (kt + 1) << 5;
        tdm_load_tile_128x32(lds_byte_off(&As[cur ^ 1][0]),
                             A + (size_t)row0 * lda + k1, (unsigned)lda,
                             (unsigned)Mp);
        tdm_load_tile_128x32(lds_byte_off(&Bs[cur ^ 1][0]),
                             B + (size_t)col0 * ldb + k1, (unsigned)ldb,
                             (unsigned)Np);
        __builtin_amdgcn_s_wait_tensorcnt((short)2);  // current tiles done
      } else {
        __builtin_amdgcn_s_wait_tensorcnt((short)0);
      }
    }
#else
    {
      const int k0 = kt << 5;
      uint4* a4d = (uint4*)&As[cur][0];
      uint4* b4d = (uint4*)&Bs[cur][0];
#pragma unroll
      for (int t = tid; t < 512; t += 256) {             // 512 data uint4/tile
        const int row = t >> 2;
        const int kq  = t & 3;
        a4d[row * 5 + kq] =
            *(const uint4*)(A + (size_t)(row0 + row) * lda + k0 + kq * 8);
        b4d[row * 5 + kq] =
            *(const uint4*)(B + (size_t)(col0 + row) * ldb + k0 + kq * 8);
      }
    }
#endif
    __syncthreads();

    // ---- fragment loads from LDS (padded stride: 5 uint4 per row) ----
    const uint4* a4 = (const uint4*)&As[cur][0];
    const uint4* b4 = (const uint4*)&Bs[cur][0];
    Frag af[4], bf[2];
#pragma unroll
    for (int i = 0; i < 4; ++i) {
      // A 16x32 frag: lane row = l16, K in [8q,8q+8) then [8q+16,8q+24)
      const int u0 = (wm * 64 + i * 16 + l16) * 5 + q;
      af[i].q[0] = a4[u0];
      af[i].q[1] = a4[u0 + 2];
    }
#pragma unroll
    for (int j = 0; j < 2; ++j) {
      // B 32x16 frag: lane col = l16, K in [8q,8q+8) then [8q+16,8q+24)
      const int u0 = (wn * 32 + j * 16 + l16) * 5 + q;
      bf[j].q[0] = b4[u0];
      bf[j].q[1] = b4[u0 + 2];
    }

#pragma unroll
    for (int i = 0; i < 4; ++i)
#pragma unroll
      for (int j = 0; j < 2; ++j)
        acc[i][j] = __builtin_amdgcn_wmma_f32_16x16x32_bf16(
            false, af[i].v, false, bf[j].v, (short)0, acc[i][j], false, false);

    __syncthreads();
  }

  // ---- epilogue: bias + relu, store bf16 or f32 ----
#pragma unroll
  for (int i = 0; i < 4; ++i) {
#pragma unroll
    for (int j = 0; j < 2; ++j) {
      const int rb = row0 + wm * 64 + i * 16 + q * 8;
      const int c  = col0 + wn * 32 + j * 16 + l16;
      if (c < N) {
        const float bv = bias ? bf2f(bias[c]) : 0.0f;
#pragma unroll
        for (int v = 0; v < 8; ++v) {
          const int r = rb + v;
          if (r < M) {
            float x = acc[i][j][v] + bv;
            if (doRelu) x = fmaxf(x, 0.0f);
            if (outBf16)
              ((uint16_t*)C)[(size_t)r * ldc + c] = f2bf(x);
            else
              ((float*)C)[(size_t)r * ldc + c] = x;
          }
        }
      }
    }
  }
}

// ---------------------------------------------------------------------------
// fp32 -> bf16 convert with zero-padding; optional transpose of the source.
//   transpose==0: dst[r, c] = (r<srcR && c<srcC) ? src[r, c] : 0
//   transpose==1: dst[r, c] = (r<srcC && c<srcR) ? src[c, r] : 0
// dst is row-major [dstR, dstC].
// ---------------------------------------------------------------------------
__global__ __launch_bounds__(256)
void cvt_pad_bf16(const float* __restrict__ src, uint16_t* __restrict__ dst,
                  int srcR, int srcC, int dstR, int dstC, int transpose) {
  const int i = blockIdx.x * 256 + threadIdx.x;
  const int total = dstR * dstC;
  if (i >= total) return;
  const int r = i / dstC;
  const int c = i - r * dstC;
  float v = 0.0f;
  if (!transpose) {
    if (r < srcR && c < srcC) v = src[(size_t)r * srcC + c];
  } else {
    if (r < srcC && c < srcR) v = src[(size_t)c * srcC + r];
  }
  dst[(size_t)r * dstC + c] = f2bf(v);
}

// ---------------------------------------------------------------------------
// Launch
// ---------------------------------------------------------------------------
extern "C" void kernel_launch(void* const* d_in, const int* in_sizes, int n_in,
                              void* d_out, int out_size, void* d_ws,
                              size_t ws_size, hipStream_t stream) {
  const float* x  = (const float*)d_in[0];   // [4096, 2048]
  const float* V1 = (const float*)d_in[1];   // [5794, 2897]
  const float* V2 = (const float*)d_in[2];   // [2897, 5794]
  float* out = (float*)d_out;                // [4096, 2048]
  char* ws = (char*)d_ws;

  constexpr int SN  = 5794, SM = 2897;
  constexpr int KpR = 2912;                  // pad 2897 -> mult of 32
  constexpr int SNp = 5888;                  // pad 5794 -> 46*128
  constexpr int BB  = 4096, DIN = 2048, DH = 4096, DOUT = 2048;

  // parameter slice offsets (elements) inside flat Vf
  constexpr size_t OFF_B1 = 8388608;   // 4096*2048
  constexpr size_t OFF_W2 = 8392704;
  constexpr size_t OFF_B2 = 25169920;
  constexpr size_t OFF_W3 = 25174016;
  constexpr size_t OFF_B3 = 33562624;

  const size_t szVpack = ((size_t)SNp * KpR * 2 + 255) & ~(size_t)255;
  const size_t szVf    = ((size_t)SN * SN * 2 + 255) & ~(size_t)255;
  const size_t oV1b = 0;                       // later reused as h1
  const size_t oV2b = oV1b + szVpack;          // later reused as h2
  const size_t oVf  = oV2b + szVpack;
  const size_t oXb  = oVf + szVf;

  uint16_t* V1b = (uint16_t*)(ws + oV1b);
  uint16_t* V2b = (uint16_t*)(ws + oV2b);
  uint16_t* Vf  = (uint16_t*)(ws + oVf);
  uint16_t* Xb  = (uint16_t*)(ws + oXb);
  uint16_t* H1  = (uint16_t*)(ws + oV1b);      // overwrite V1b after recon
  uint16_t* H2  = (uint16_t*)(ws + oV2b);      // overwrite V2b after layer 1

  // Phase 0: pack inputs to bf16 (pad + transpose V2)
  {
    const int tot = SNp * KpR;
    cvt_pad_bf16<<<(tot + 255) / 256, 256, 0, stream>>>(V1, V1b, SN, SM, SNp,
                                                        KpR, 0);
    cvt_pad_bf16<<<(tot + 255) / 256, 256, 0, stream>>>(V2, V2b, SM, SN, SNp,
                                                        KpR, 1);
    const int totx = BB * DIN;
    cvt_pad_bf16<<<(totx + 255) / 256, 256, 0, stream>>>(x, Xb, BB, DIN, BB,
                                                         DIN, 0);
  }

  // Phase 1: Vf = V1 @ V2  (flat [SN*SN] bf16)
  gemm_bf16_wmma<<<dim3(46, 46), 256, 0, stream>>>(
      V1b, KpR, SNp, V2b, KpR, SNp, (void*)Vf, SN, SN, SN, KpR, nullptr, 0, 1);

  // Phase 2: h1 = relu(x @ W1^T + b1)
  gemm_bf16_wmma<<<dim3(32, 32), 256, 0, stream>>>(
      Xb, DIN, BB, Vf, DIN, DH, (void*)H1, DH, BB, DH, DIN, Vf + OFF_B1, 1, 1);

  // Phase 3: h2 = relu(h1 @ W2^T + b2)
  gemm_bf16_wmma<<<dim3(32, 32), 256, 0, stream>>>(
      H1, DH, BB, Vf + OFF_W2, DH, DH, (void*)H2, DH, BB, DH, DH, Vf + OFF_B2,
      1, 1);

  // Phase 4: out = h2 @ W3^T + b3   (fp32 out)
  gemm_bf16_wmma<<<dim3(16, 32), 256, 0, stream>>>(
      H2, DH, BB, Vf + OFF_W3, DH, DOUT, (void*)out, DOUT, BB, DOUT, DH,
      Vf + OFF_B3, 0, 0);

  (void)in_sizes; (void)n_in; (void)out_size; (void)ws_size;
}
